// CrossAttention_1778116460878
// MI455X (gfx1250) — compile-verified, matrix operand order checked by reading, not run on hardware
//
#include <hip/hip_runtime.h>

// ---------------------------------------------------------------------------
// CDNA5 (gfx1250) cross-attention, bf16 WMMA (v_wmma_f32_16x16x32_bf16),
// all WMMA fragments loaded as contiguous b128 pairs, async global->LDS
// staging via __builtin_amdgcn_global_load_async_to_lds_b128.
// ---------------------------------------------------------------------------

typedef __bf16 bf16;
typedef __attribute__((ext_vector_type(16))) __bf16 v16bf;
typedef __attribute__((ext_vector_type(8)))  __bf16 v8bf;
typedef __attribute__((ext_vector_type(8)))  float  v8f;
typedef __attribute__((ext_vector_type(4)))  int    v4i;

#if __has_builtin(__builtin_amdgcn_global_load_async_to_lds_b128)
#define CA_ASYNC_LDS 1
#else
#define CA_ASYNC_LDS 0
#endif

// Copy 16 bytes global -> LDS (async when available).
__device__ __forceinline__ void copy_b128_to_lds(bf16* l, const bf16* g) {
#if CA_ASYNC_LDS
    __builtin_amdgcn_global_load_async_to_lds_b128(
        (__attribute__((address_space(1))) v4i*)g,
        (__attribute__((address_space(3))) v4i*)l, 0, 0);
#else
    *(v8bf*)l = *(const v8bf*)g;
#endif
}

__device__ __forceinline__ void wait_lds_copies() {
#if CA_ASYNC_LDS
#if __has_builtin(__builtin_amdgcn_s_wait_asynccnt)
    __builtin_amdgcn_s_wait_asynccnt(0);
#else
    asm volatile("s_wait_asynccnt 0x0" ::: "memory");
#endif
#endif
}

// 16-bit WMMA A/B fragment: element e of lane maps to reduction index
//   kbase(lane) + (e>=8 ? 16 : 0) + (e&7)
// lane<16 -> kbase 0 (K 0..7,16..23); lane>=16 -> kbase 8 (K 8..15,24..31).
// When the per-lane data is contiguous along the reduction dim this is two
// 16-byte loads.
__device__ __forceinline__ v16bf frag_contig(const bf16* p, int lane) {
    const int kb = (lane & 16) ? 8 : 0;
    v8bf lo = *(const v8bf*)(p + kb);
    v8bf hi = *(const v8bf*)(p + kb + 16);
    v16bf r;
#pragma unroll
    for (int i = 0; i < 8; ++i) { r[i] = lo[i]; r[i + 8] = hi[i]; }
    return r;
}

__device__ __forceinline__ v8f wmma_bf16(v16bf a, v16bf b, v8f c) {
    return __builtin_amdgcn_wmma_f32_16x16x32_bf16(false, a, false, b, (short)0, c,
                                                   false, false);
}

// ---------------------------------------------------------------------------
// LayerNorm (f32 in) -> bf16 out.  One row per 256-thread block.
// ---------------------------------------------------------------------------
__global__ void ca_ln_bf16(const float* __restrict__ x, const float* __restrict__ g,
                           const float* __restrict__ b, bf16* __restrict__ out, int cols) {
    __shared__ float red[256];
    const int row = blockIdx.x;
    const float* xr = x + (size_t)row * cols;
    float s = 0.f, s2 = 0.f;
    for (int c = threadIdx.x; c < cols; c += 256) {
        float v = xr[c];
        s += v;
        s2 += v * v;
    }
    red[threadIdx.x] = s;
    __syncthreads();
    for (int o = 128; o > 0; o >>= 1) {
        if (threadIdx.x < o) red[threadIdx.x] += red[threadIdx.x + o];
        __syncthreads();
    }
    const float mean = red[0] / cols;
    __syncthreads();
    red[threadIdx.x] = s2;
    __syncthreads();
    for (int o = 128; o > 0; o >>= 1) {
        if (threadIdx.x < o) red[threadIdx.x] += red[threadIdx.x + o];
        __syncthreads();
    }
    const float var = red[0] / cols - mean * mean;
    const float rstd = rsqrtf(var + 1e-5f);
    bf16* orow = out + (size_t)row * cols;
    for (int c = threadIdx.x; c < cols; c += 256)
        orow[c] = (bf16)((xr[c] - mean) * rstd * g[c] + b[c]);
}

// ---------------------------------------------------------------------------
// Tiled transpose + f32 -> bf16: in[K][N] -> out[N][K].
// ---------------------------------------------------------------------------
__global__ void ca_transpose_bf16(const float* __restrict__ in, bf16* __restrict__ out,
                                  int K, int N) {
    __shared__ float t[32][33];
    const int k0 = blockIdx.y * 32;
    const int n0 = blockIdx.x * 32;
#pragma unroll
    for (int r = threadIdx.y; r < 32; r += 8)
        t[r][threadIdx.x] = in[(size_t)(k0 + r) * N + n0 + threadIdx.x];
    __syncthreads();
#pragma unroll
    for (int r = threadIdx.y; r < 32; r += 8)
        out[(size_t)(n0 + r) * K + k0 + threadIdx.x] = (bf16)t[threadIdx.x][r];
}

// ---------------------------------------------------------------------------
// C[M,N] = A[M,K] @ B[K,N] with B supplied TRANSPOSED (Bt[N][K]).
// Block = 256 threads = 8 waves; wave tile 16x16; block tile 32(M) x 64(N).
// A (32x32) and Bt (64x32) tiles staged in LDS; all fragments contiguous.
// MODE: 0 = f32 row-major, 1 = bf16 row-major,
//       2 = bf16 transposed per-head (Vt[b,h,d,j], seq = 2048, head = 64).
// ---------------------------------------------------------------------------
template <int MODE>
__global__ void ca_gemm_bf16(const bf16* __restrict__ A, const bf16* __restrict__ Bt,
                             void* __restrict__ C, int M, int N, int K) {
    __shared__ __align__(16) bf16 sA[32 * 32];
    __shared__ __align__(16) bf16 sB[64 * 32];
    const int lane = threadIdx.x & 31;
    const int wave = threadIdx.x >> 5;
    const int mt = wave & 1;
    const int nt = wave >> 1;
    const int m0blk = blockIdx.y * 32;
    const int m0 = m0blk + mt * 16;
    const int nblk = blockIdx.x * 64;

    v8f acc = {};
    for (int k0 = 0; k0 < K; k0 += 32) {
        {  // Bt tile: 64 rows (n) x 32 (k); 8 elems (16B) per thread.
            const int idx = threadIdx.x * 8;
            const int row = idx >> 5;
            const int col = idx & 31;
            copy_b128_to_lds(sB + idx, Bt + (size_t)(nblk + row) * K + k0 + col);
        }
        if (threadIdx.x < 128) {  // A tile: 32 rows (m) x 32 (k).
            const int idx = threadIdx.x * 8;
            const int row = idx >> 5;
            const int col = idx & 31;
            copy_b128_to_lds(sA + idx, A + (size_t)(m0blk + row) * K + k0 + col);
        }
        if (k0 + 32 < K)
            __builtin_prefetch(Bt + (size_t)(nblk + (threadIdx.x >> 2)) * K + k0 + 32, 0, 1);
        wait_lds_copies();
        __syncthreads();
        v16bf af = frag_contig(sA + (size_t)(mt * 16 + (lane & 15)) * 32, lane);
        v16bf bfr = frag_contig(sB + (size_t)(nt * 16 + (lane & 15)) * 32, lane);
        acc = wmma_bf16(af, bfr, acc);
        __syncthreads();
    }

    const int ncol = nblk + nt * 16 + (lane & 15);
    const int rbase = m0 + ((lane & 16) ? 8 : 0);
    if (MODE == 2) {
        // Lane owns column ncol=(h,d) and 8 consecutive rows m=j: one b128
        // store into Vt[((b*16+h)*64+d) * 2048 + j].
        const int bidx = rbase >> 11;  // m / 2048
        const int j = rbase & 2047;
        const int h = ncol >> 6;
        const int d = ncol & 63;
        v8bf pk;
#pragma unroll
        for (int r = 0; r < 8; ++r) pk[r] = (bf16)acc[r];
        *(v8bf*)((bf16*)C + ((size_t)((bidx * 16 + h) * 64 + d)) * 2048 + j) = pk;
    } else {
#pragma unroll
        for (int r = 0; r < 8; ++r) {
            const size_t off = (size_t)(rbase + r) * N + ncol;
            if (MODE == 1)
                ((bf16*)C)[off] = (bf16)acc[r];
            else
                ((float*)C)[off] = acc[r];
        }
    }
}

// ---------------------------------------------------------------------------
// Flash attention, one 16-row Q tile per wave; 4 waves/block share (b,h) and
// LDS-staged K (32x64, row-major j,d) and Vt (64x32, row-major d,j) chunks.
// Computes S^T = K * Q^T so the softmaxed probabilities are already in
// A-operand layout for P @ V (no cross-lane movement).
// ---------------------------------------------------------------------------
__global__ void ca_flash_attn(const bf16* __restrict__ Q, const bf16* __restrict__ Kp,
                              const bf16* __restrict__ Vt, bf16* __restrict__ O,
                              int seq_q, int seq_k) {
    __shared__ __align__(16) bf16 sK[32 * 64];
    __shared__ __align__(16) bf16 sVt[64 * 32];
    const int lane = threadIdx.x & 31;
    const int wave = threadIdx.x >> 5;
    const int bi = blockIdx.x;
    const int TPB = 32;  // i-tiles per (b,h) / waves-per-block = 128/4
    const int b = bi / (16 * TPB);
    const int h = (bi / TPB) % 16;
    const int i0 = ((bi % TPB) * 4 + wave) * 16;
    const size_t ld = 1024;  // INNER

    const bf16* Qb = Q + ((size_t)b * seq_q) * ld + (size_t)h * 64;
    const bf16* Kb = Kp + ((size_t)b * seq_k) * ld + (size_t)h * 64;
    const bf16* VtB = Vt + (size_t)((b * 16 + h) * 64) * seq_k;

    // Q^T as B operand: lane holds column i = i0 + (lane&15); reduction dim d
    // contiguous in Q's row.
    const bf16* qr = Qb + (size_t)(i0 + (lane & 15)) * ld;
    const v16bf qb0 = frag_contig(qr, lane);       // d = 0..31
    const v16bf qb1 = frag_contig(qr + 32, lane);  // d = 32..63

    float m_run = -1e30f, l_run = 0.f;
    v8f o0 = {}, o1 = {}, o2 = {}, o3 = {};
    const float scale = 0.125f;  // 1/sqrt(64)

    for (int j0 = 0; j0 < seq_k; j0 += 32) {
        {  // K chunk: 32 rows (j) x 64 (d); Vt chunk: 64 rows (d) x 32 (j).
            const int idx = threadIdx.x * 16;
            const int jr = idx >> 6;
            const int dc = idx & 63;
            copy_b128_to_lds(sK + idx, Kb + (size_t)(j0 + jr) * ld + dc);
            copy_b128_to_lds(sK + idx + 8, Kb + (size_t)(j0 + jr) * ld + dc + 8);
            const int vr = threadIdx.x >> 1;        // d row 0..63
            const int vc = (threadIdx.x & 1) * 16;  // j col 0 or 16
            const int vidx = vr * 32 + vc;
            copy_b128_to_lds(sVt + vidx, VtB + (size_t)vr * seq_k + j0 + vc);
            copy_b128_to_lds(sVt + vidx + 8, VtB + (size_t)vr * seq_k + j0 + vc + 8);
        }
        wait_lds_copies();
        __syncthreads();

        // S^T tiles: A = K rows (M=j, reduction d), B = Q^T.
        const int jl = lane & 15;
        v8f s0 = {}, s1 = {};
        s0 = wmma_bf16(frag_contig(sK + (size_t)jl * 64, lane), qb0, s0);
        s0 = wmma_bf16(frag_contig(sK + (size_t)jl * 64 + 32, lane), qb1, s0);
        s1 = wmma_bf16(frag_contig(sK + (size_t)(16 + jl) * 64, lane), qb0, s1);
        s1 = wmma_bf16(frag_contig(sK + (size_t)(16 + jl) * 64 + 32, lane), qb1, s1);

        // Online softmax over j for column i = lane&15 (stats split across the
        // two wave halves; combine with shfl_xor 16).
        float mloc = -1e30f;
#pragma unroll
        for (int r = 0; r < 8; ++r) {
            s0[r] *= scale;
            s1[r] *= scale;
            mloc = fmaxf(mloc, fmaxf(s0[r], s1[r]));
        }
        mloc = fmaxf(mloc, __shfl_xor(mloc, 16, 32));
        const float m_new = fmaxf(m_run, mloc);
        const float alpha = __expf(m_run - m_new);
        float psum = 0.f;
        v16bf pa;
#pragma unroll
        for (int r = 0; r < 8; ++r) {
            float p0 = __expf(s0[r] - m_new);
            float p1 = __expf(s1[r] - m_new);
            psum += p0 + p1;
            pa[r] = (bf16)p0;      // K(j) = kbase + r        (tile 0)
            pa[r + 8] = (bf16)p1;  // K(j) = kbase + 16 + r   (tile 1)
        }
        psum += __shfl_xor(psum, 16, 32);
        l_run = l_run * alpha + psum;
        m_run = m_new;

        // Rescale running O by per-row alpha (row M=r lives at lane r / r+8).
        {
            const int sb = (lane & 16) ? 8 : 0;
#pragma unroll
            for (int r = 0; r < 8; ++r) {
                const float ar = __shfl(alpha, sb + r, 32);
                o0[r] *= ar;
                o1[r] *= ar;
                o2[r] *= ar;
                o3[r] *= ar;
            }
        }

        // O += P @ V: B fragments from Vt rows (reduction dim j contiguous).
        o0 = wmma_bf16(pa, frag_contig(sVt + (size_t)(0 + (lane & 15)) * 32, lane), o0);
        o1 = wmma_bf16(pa, frag_contig(sVt + (size_t)(16 + (lane & 15)) * 32, lane), o1);
        o2 = wmma_bf16(pa, frag_contig(sVt + (size_t)(32 + (lane & 15)) * 32, lane), o2);
        o3 = wmma_bf16(pa, frag_contig(sVt + (size_t)(48 + (lane & 15)) * 32, lane), o3);
        __syncthreads();
    }

    // Normalize by l and store bf16 attention output (row-major, per-head).
    const int sb = (lane & 16) ? 8 : 0;
    const int nloc = lane & 15;
#pragma unroll
    for (int r = 0; r < 8; ++r) {
        const float linv = 1.f / __shfl(l_run, sb + r, 32);
        const size_t off = ((size_t)b * seq_q + (i0 + sb + r)) * ld + (size_t)h * 64;
        O[off + 0 + nloc] = (bf16)(o0[r] * linv);
        O[off + 16 + nloc] = (bf16)(o1[r] * linv);
        O[off + 32 + nloc] = (bf16)(o2[r] * linv);
        O[off + 48 + nloc] = (bf16)(o3[r] * linv);
    }
}

// ---------------------------------------------------------------------------
// Host-side orchestration.
// ---------------------------------------------------------------------------
extern "C" void kernel_launch(void* const* d_in, const int* in_sizes, int n_in,
                              void* d_out, int out_size, void* d_ws, size_t ws_size,
                              hipStream_t stream) {
    (void)in_sizes;
    (void)n_in;
    (void)out_size;
    (void)ws_size;

    const float* x = (const float*)d_in[0];
    const float* ctx = (const float*)d_in[1];
    const float* ng = (const float*)d_in[2];
    const float* nb = (const float*)d_in[3];
    const float* ncg = (const float*)d_in[4];
    const float* ncb = (const float*)d_in[5];
    const float* Wq = (const float*)d_in[6];
    const float* Wk = (const float*)d_in[7];
    const float* Wv = (const float*)d_in[8];
    const float* Wo = (const float*)d_in[9];
    float* out = (float*)d_out;

    const int B = 4, NQ = 2048, NK = 2048, DIM = 1024, CDIM = 768, INNER = 1024;
    const int MQ = B * NQ;  // 8192 query rows
    const int MK = B * NK;  // 8192 context rows

    size_t off = 0;
    auto alloc = [&](size_t elems) {
        bf16* p = (bf16*)((char*)d_ws + off);
        off += ((elems * sizeof(bf16) + 255) / 256) * 256;
        return p;
    };
    bf16* xn = alloc((size_t)MQ * DIM);
    bf16* cn = alloc((size_t)MK * CDIM);
    bf16* wqT = alloc((size_t)DIM * INNER);   // [n][k] = [INNER][DIM]
    bf16* wkT = alloc((size_t)CDIM * INNER);  // [INNER][CDIM]
    bf16* wvT = alloc((size_t)CDIM * INNER);  // [INNER][CDIM]
    bf16* woT = alloc((size_t)INNER * DIM);   // [DIM][INNER]
    bf16* Qp = alloc((size_t)MQ * INNER);     // row-major (b*n, h*d)
    bf16* Kp = alloc((size_t)MK * INNER);     // row-major (b*n, h*d)
    bf16* Vtp = alloc((size_t)MK * INNER);    // transposed per-head (b,h,d,j)
    bf16* AO = alloc((size_t)MQ * INNER);

    // LayerNorms -> bf16 activations.
    ca_ln_bf16<<<MQ, 256, 0, stream>>>(x, ng, nb, xn, DIM);
    ca_ln_bf16<<<MK, 256, 0, stream>>>(ctx, ncg, ncb, cn, CDIM);

    // Weight transpose + conversion: W[k][n] -> Wt[n][k] bf16.
    dim3 tb(32, 8);
    ca_transpose_bf16<<<dim3(INNER / 32, DIM / 32), tb, 0, stream>>>(Wq, wqT, DIM, INNER);
    ca_transpose_bf16<<<dim3(INNER / 32, CDIM / 32), tb, 0, stream>>>(Wk, wkT, CDIM, INNER);
    ca_transpose_bf16<<<dim3(INNER / 32, CDIM / 32), tb, 0, stream>>>(Wv, wvT, CDIM, INNER);
    ca_transpose_bf16<<<dim3(DIM / 32, INNER / 32), tb, 0, stream>>>(Wo, woT, INNER, DIM);

    // Projections.
    ca_gemm_bf16<1><<<dim3(INNER / 64, MQ / 32), 256, 0, stream>>>(xn, wqT, (void*)Qp,
                                                                   MQ, INNER, DIM);
    ca_gemm_bf16<1><<<dim3(INNER / 64, MK / 32), 256, 0, stream>>>(cn, wkT, (void*)Kp,
                                                                   MK, INNER, CDIM);
    ca_gemm_bf16<2><<<dim3(INNER / 64, MK / 32), 256, 0, stream>>>(cn, wvT, (void*)Vtp,
                                                                   MK, INNER, CDIM);

    // Flash attention: 4 b * 16 h * 128 i-tiles, 4 waves (i-tiles) per block.
    ca_flash_attn<<<B * 16 * 32, 128, 0, stream>>>(Qp, Kp, Vtp, AO, NQ, NK);

    // Output projection (f32 out -> d_out).
    ca_gemm_bf16<0><<<dim3(DIM / 64, MQ / 32), 256, 0, stream>>>(AO, woT, (void*)out,
                                                                 MQ, DIM, INNER);
}
